// GPT_30975304139023
// MI455X (gfx1250) — matrix operand order repaired and checked by reading, not compiled, not run on hardware
//
#include <hip/hip_runtime.h>
#include <hip/hip_bf16.h>

typedef __attribute__((ext_vector_type(16))) __bf16 v16bf;
typedef __attribute__((ext_vector_type(8)))  float  v8f;

#define Lc 4
#define Dc 768
#define Hc 12
#define HDc 64
#define Vc 50257
#define Tc 2048
#define D3c 2304
#define D4c 3072

__device__ __forceinline__ v8f wmma_bf16(v16bf a, v16bf b, v8f c) {
  return __builtin_amdgcn_wmma_f32_16x16x32_bf16(false, a, false, b, (short)0, c,
                                                 false, false);
}

// 16 contiguous bf16 -> fragment (two 16B loads)
__device__ __forceinline__ v16bf load_frag16(const __bf16* __restrict__ p) {
  union { v16bf v; uint4 q[2]; } u;
  u.q[0] = *reinterpret_cast<const uint4*>(p);
  u.q[1] = *reinterpret_cast<const uint4*>(p + 8);
  return u.v;
}

// A-fragment (16x32 bf16), A row-major [M][K], tile (m0,k0).
// Lane: row=lane&15, hi=lane>>4; e<8 -> K=k0+8*hi+e ; e>=8 -> K=k0+16+8*hi+(e-8)
__device__ __forceinline__ v16bf load_a_frag(const __bf16* __restrict__ A, int lda,
                                             int m0, int k0, int lane) {
  int row = lane & 15, hi = lane >> 4;
  const __bf16* p = A + (size_t)(m0 + row) * lda + k0 + 8 * hi;
  union { v16bf v; uint4 q[2]; } u;
  u.q[0] = *reinterpret_cast<const uint4*>(p);
  u.q[1] = *reinterpret_cast<const uint4*>(p + 16);
  return u.v;
}

// B-fragment (32x16) from transposed storage Bt[N][K]: e<8 -> Bt[n][k0+8*hi+e] ; e>=8 -> Bt[n][k0+16+8*hi+(e-8)]
__device__ __forceinline__ v16bf load_b_frag_nk(const __bf16* __restrict__ Bt, int ldb,
                                                int k0, int n, int lane) {
  int hi = lane >> 4;
  const __bf16* p = Bt + (size_t)n * ldb + k0 + 8 * hi;
  union { v16bf v; uint4 q[2]; } u;
  u.q[0] = *reinterpret_cast<const uint4*>(p);
  u.q[1] = *reinterpret_cast<const uint4*>(p + 16);
  return u.v;
}

// B-fragment from [K][N] storage (strided; only used for V in attention)
__device__ __forceinline__ v16bf load_b_frag_kn(const __bf16* __restrict__ B, int ldb,
                                                int k0, int n, int lane) {
  int hi = lane >> 4;
  const __bf16* p = B + (size_t)(k0 + 8 * hi) * ldb + n;
  union { v16bf v; __bf16 e[16]; } u;
#pragma unroll
  for (int i = 0; i < 16; ++i) u.e[i] = p[(size_t)(i < 8 ? i : i + 8) * ldb];
  return u.v;
}

// One async-to-LDS instruction per wave: block stages A[Mb..Mb+63][k0..k0+31] (4KB).
// Thread tid copies 16B: row = tid>>2, chunk = tid&3.
__device__ __forceinline__ void stage_a_async(const __bf16* __restrict__ A, int lda,
                                              int Mb, int k0, __bf16 (*buf)[32],
                                              int tid) {
  int row = tid >> 2, chunk = tid & 3;
  const __bf16* g = A + (size_t)(Mb + row) * lda + k0 + chunk * 8;
  unsigned lds_off = (unsigned)(uintptr_t)(&buf[row][chunk * 8]);
  unsigned long long ga = (unsigned long long)(uintptr_t)g;
  asm volatile("global_load_async_to_lds_b128 %0, %1, off"
               :
               : "v"(lds_off), "v"(ga)
               : "memory");
}

// C = A[MxK] @ Bt[NxK]^T (+bias)(+GELU)(+resid).
// Block (8 waves) = 64(M) x 256(N) macrotile; wave (wm,wn) = 32x64 tile.
// A k-slices double-buffered in LDS via async-to-LDS; B direct from L2.
template <bool GELU_ACT, bool RESID, bool OUTBF>
__global__ __launch_bounds__(256) void gemm_wmma(
    const __bf16* __restrict__ A, const __bf16* __restrict__ Bt,
    const float* __restrict__ bias, const float* __restrict__ resid,
    void* __restrict__ Cout, int M, int N, int K, int lda, int ldc) {
  __shared__ __bf16 Alds[2][64][32];
  int tid = threadIdx.x;
  int lane = tid & 31, wid = tid >> 5;
  int wm = wid >> 2, wn = wid & 3;
  int Mb = blockIdx.y << 6;
  int m0 = Mb + (wm << 5);
  int n0 = (blockIdx.x << 8) + (wn << 6);
  int col = lane & 15, hi = lane >> 4;

  v8f acc[2][4] = {};
  int nsteps = K >> 5;
  stage_a_async(A, lda, Mb, 0, Alds[0], tid);
  for (int s = 0; s < nsteps; ++s) {
    const __bf16(*buf)[32] = Alds[s & 1];
    if (s + 1 < nsteps) {
      stage_a_async(A, lda, Mb, (s + 1) << 5, Alds[(s + 1) & 1], tid);
      asm volatile("s_wait_asynccnt 1" ::: "memory");
    } else {
      asm volatile("s_wait_asynccnt 0" ::: "memory");
    }
    __syncthreads();  // slice s visible to all waves
    // A fragments from LDS (ds_load_b128 x4)
    v16bf a0, a1;
    {
      const __bf16* r0 = &buf[(wm << 5) + col][0];
      const __bf16* r1 = &buf[(wm << 5) + 16 + col][0];
      union { v16bf v; uint4 q[2]; } u0, u1;
      u0.q[0] = *(const uint4*)(r0 + 8 * hi);
      u0.q[1] = *(const uint4*)(r0 + 16 + 8 * hi);
      u1.q[0] = *(const uint4*)(r1 + 8 * hi);
      u1.q[1] = *(const uint4*)(r1 + 16 + 8 * hi);
      a0 = u0.v; a1 = u1.v;
    }
    int k0 = s << 5;
#pragma unroll
    for (int j = 0; j < 4; ++j) {
      int n = n0 + 16 * j + col;
      int nc = n < N ? n : N - 1;  // clamp loads; stores predicated below
      v16bf b = load_b_frag_nk(Bt, K, k0, nc, lane);
      acc[0][j] = wmma_bf16(a0, b, acc[0][j]);
      acc[1][j] = wmma_bf16(a1, b, acc[1][j]);
    }
    __syncthreads();  // protect buf before it is overwritten by stage s+2
  }
#pragma unroll
  for (int j = 0; j < 4; ++j) {
    int n = n0 + 16 * j + col;
    if (n < N) {
      float bv = bias ? bias[n] : 0.0f;
#pragma unroll
      for (int i = 0; i < 2; ++i) {
#pragma unroll
        for (int r = 0; r < 8; ++r) {
          int m = m0 + 16 * i + r + 8 * hi;
          float v = acc[i][j][r] + bv;
          if (GELU_ACT) v = 0.5f * v * (1.0f + erff(v * 0.70710678118654752f));
          if (RESID) v += resid[(size_t)m * ldc + n];
          size_t idx = (size_t)m * ldc + n;
          if (OUTBF) ((__bf16*)Cout)[idx] = (__bf16)v;
          else       ((float*)Cout)[idx] = v;
        }
      }
    }
  }
}

// Flash attention: one wave per (16-query tile, head). qkv: bf16 [T][3D], y: bf16 [T][D]
__global__ __launch_bounds__(32) void attention_kernel(
    const __bf16* __restrict__ qkv, __bf16* __restrict__ y) {
  __shared__ __bf16 p_lds[16][32];
  int lane = threadIdx.x;
  int q0 = blockIdx.x << 4;
  int h  = blockIdx.y;
  int col = lane & 15, hi = lane >> 4;

  const __bf16* Qb = qkv + h * HDc;
  const __bf16* Kb = qkv + Dc + h * HDc;
  const __bf16* Vb = qkv + 2 * Dc + h * HDc;

  v16bf aq[2];
  aq[0] = load_a_frag(Qb, D3c, q0, 0, lane);
  aq[1] = load_a_frag(Qb, D3c, q0, 32, lane);

  float m_run[8], l_run[8];
#pragma unroll
  for (int r = 0; r < 8; ++r) { m_run[r] = -1.0e30f; l_run[r] = 0.0f; }
  v8f o[4] = {};

  int kend = q0 + 16;
  for (int kb = 0; kb < kend; kb += 32) {
    v8f s[2] = {};
#pragma unroll
    for (int sub = 0; sub < 2; ++sub) {
      int key = kb + 16 * sub + col;
      int keyc = key < Tc ? key : Tc - 1;
#pragma unroll
      for (int kk = 0; kk < 2; ++kk) {
        v16bf b = load_b_frag_nk(Kb, D3c, kk * 32, keyc, lane);
        s[sub] = wmma_bf16(aq[kk], b, s[sub]);
      }
    }
#pragma unroll
    for (int sub = 0; sub < 2; ++sub) {
      int key = kb + 16 * sub + col;
#pragma unroll
      for (int r = 0; r < 8; ++r) {
        int qrow = q0 + r + 8 * hi;
        float v = s[sub][r] * 0.125f;  // 1/sqrt(64)
        s[sub][r] = (key <= qrow) ? v : -1.0e30f;
      }
    }
#pragma unroll
    for (int r = 0; r < 8; ++r) {
      float mx = fmaxf(s[0][r], s[1][r]);
#pragma unroll
      for (int off = 8; off; off >>= 1) mx = fmaxf(mx, __shfl_xor(mx, off, 32));
      float mnew = fmaxf(m_run[r], mx);
      float alpha = __expf(m_run[r] - mnew);
      float p0 = __expf(s[0][r] - mnew);
      float p1 = __expf(s[1][r] - mnew);
      s[0][r] = p0; s[1][r] = p1;
      float rs = p0 + p1;
#pragma unroll
      for (int off = 8; off; off >>= 1) rs += __shfl_xor(rs, off, 32);
      l_run[r] = l_run[r] * alpha + rs;
      m_run[r] = mnew;
#pragma unroll
      for (int j = 0; j < 4; ++j) o[j][r] *= alpha;
    }
#pragma unroll
    for (int sub = 0; sub < 2; ++sub)
#pragma unroll
      for (int r = 0; r < 8; ++r)
        p_lds[r + 8 * hi][16 * sub + col] = (__bf16)s[sub][r];
    __syncthreads();
    v16bf pa;
    {
      union { v16bf v; __bf16 e[16]; } u;
      int row = col;
#pragma unroll
      for (int i = 0; i < 8; ++i) u.e[i] = p_lds[row][i + 8 * hi];
#pragma unroll
      for (int i = 0; i < 8; ++i) u.e[8 + i] = p_lds[row][16 + i + 8 * hi];
      pa = u.v;
    }
    __syncthreads();
#pragma unroll
    for (int j = 0; j < 4; ++j) {
      v16bf b = load_b_frag_kn(Vb, D3c, kb, 16 * j + col, lane);
      o[j] = wmma_bf16(pa, b, o[j]);
    }
  }
#pragma unroll
  for (int j = 0; j < 4; ++j) {
    int dim = 16 * j + col;
#pragma unroll
    for (int r = 0; r < 8; ++r) {
      int q = q0 + r + 8 * hi;
      y[(size_t)q * Dc + h * HDc + dim] = (__bf16)(o[j][r] / l_run[r]);
    }
  }
}

// w[K][N] f32 -> wt[N][K] bf16 (LDS-tiled 32x32 transpose)
__global__ __launch_bounds__(256) void transpose_bf16_kernel(
    const float* __restrict__ w, __bf16* __restrict__ wt, int K, int N) {
  __shared__ float tile[32][33];
  int nb = blockIdx.x << 5, kb = blockIdx.y << 5;
  int tx = threadIdx.x & 31, ty = threadIdx.x >> 5;
#pragma unroll
  for (int i = ty; i < 32; i += 8) tile[i][tx] = w[(size_t)(kb + i) * N + nb + tx];
  __syncthreads();
#pragma unroll
  for (int i = ty; i < 32; i += 8)
    wt[(size_t)(nb + i) * K + kb + tx] = (__bf16)tile[tx][i];
}

__global__ __launch_bounds__(256) void cvt_bf16_kernel(
    const float* __restrict__ in, __bf16* __restrict__ out, int n) {
  int i = blockIdx.x * 256 + threadIdx.x;
  if (i < n) out[i] = (__bf16)in[i];
}

__global__ __launch_bounds__(256) void embed_kernel(
    const int* __restrict__ x, const float* __restrict__ wte,
    const float* __restrict__ wpe, float* __restrict__ h) {
  int t = blockIdx.x;
  int tok = x[t];
  for (int i = threadIdx.x; i < Dc; i += 256)
    h[(size_t)t * Dc + i] = wte[(size_t)tok * Dc + i] + wpe[(size_t)t * Dc + i];
}

// f32 in, bf16 out
__global__ __launch_bounds__(256) void layernorm_kernel(
    const float* __restrict__ x, const float* __restrict__ w,
    const float* __restrict__ b, __bf16* __restrict__ out) {
  int t = blockIdx.x;
  const float* row = x + (size_t)t * Dc;
  __shared__ float shs[8], shq[8];
  __shared__ float mean_s, rstd_s;
  float s = 0.0f, ss = 0.0f;
  for (int i = threadIdx.x; i < Dc; i += 256) {
    float v = row[i]; s += v; ss += v * v;
  }
#pragma unroll
  for (int off = 16; off; off >>= 1) {
    s += __shfl_xor(s, off, 32);
    ss += __shfl_xor(ss, off, 32);
  }
  int lane = threadIdx.x & 31, wid = threadIdx.x >> 5;
  if (lane == 0) { shs[wid] = s; shq[wid] = ss; }
  __syncthreads();
  if (threadIdx.x == 0) {
    float S = 0.0f, Q = 0.0f;
#pragma unroll
    for (int i = 0; i < 8; ++i) { S += shs[i]; Q += shq[i]; }
    float mean = S * (1.0f / Dc);
    float var = Q * (1.0f / Dc) - mean * mean;
    mean_s = mean; rstd_s = rsqrtf(var + 1e-5f);
  }
  __syncthreads();
  float mean = mean_s, rstd = rstd_s;
  for (int i = threadIdx.x; i < Dc; i += 256)
    out[(size_t)t * Dc + i] = (__bf16)((row[i] - mean) * rstd * w[i] + b[i]);
}

__global__ __launch_bounds__(256) void nll_kernel(
    const float* __restrict__ logits, const int* __restrict__ tgt,
    float* __restrict__ nll, float* __restrict__ vld) {
  int t = blockIdx.x;
  const float* row = logits + (size_t)t * Vc;
  __shared__ float sh[8];
  int lane = threadIdx.x & 31, wid = threadIdx.x >> 5;
  float mx = -3.0e38f;
  for (int i = threadIdx.x; i < Vc; i += 256) mx = fmaxf(mx, row[i]);
#pragma unroll
  for (int off = 16; off; off >>= 1) mx = fmaxf(mx, __shfl_xor(mx, off, 32));
  if (lane == 0) sh[wid] = mx;
  __syncthreads();
  float m_all = sh[0];
#pragma unroll
  for (int i = 1; i < 8; ++i) m_all = fmaxf(m_all, sh[i]);
  __syncthreads();
  float se = 0.0f;
  for (int i = threadIdx.x; i < Vc; i += 256) se += __expf(row[i] - m_all);
#pragma unroll
  for (int off = 16; off; off >>= 1) se += __shfl_xor(se, off, 32);
  if (lane == 0) sh[wid] = se;
  __syncthreads();
  if (threadIdx.x == 0) {
    float Z = 0.0f;
#pragma unroll
    for (int i = 0; i < 8; ++i) Z += sh[i];
    int tg = tgt[t];
    bool ok = (tg != -1);
    int ti = ok ? tg : 0;
    float lp = row[ti] - m_all - logf(Z);
    nll[t] = ok ? -lp : 0.0f;
    vld[t] = ok ? 1.0f : 0.0f;
  }
}

__global__ __launch_bounds__(256) void loss_reduce_kernel(
    const float* __restrict__ nll, const float* __restrict__ vld,
    float* __restrict__ out) {
  __shared__ float sh[16];
  float s = 0.0f, c = 0.0f;
  for (int i = threadIdx.x; i < Tc; i += 256) { s += nll[i]; c += vld[i]; }
#pragma unroll
  for (int off = 16; off; off >>= 1) {
    s += __shfl_xor(s, off, 32);
    c += __shfl_xor(c, off, 32);
  }
  int lane = threadIdx.x & 31, wid = threadIdx.x >> 5;
  if (lane == 0) { sh[wid] = s; sh[8 + wid] = c; }
  __syncthreads();
  if (threadIdx.x == 0) {
    float S = 0.0f, C = 0.0f;
#pragma unroll
    for (int i = 0; i < 8; ++i) { S += sh[i]; C += sh[8 + i]; }
    out[0] = S / fmaxf(C, 1.0f);
  }
}

static inline dim3 gemm_grid(int M, int N) {
  return dim3((N + 255) / 256, M / 64);
}

extern "C" void kernel_launch(void* const* d_in, const int* in_sizes, int n_in,
                              void* d_out, int out_size, void* d_ws, size_t ws_size,
                              hipStream_t stream) {
  const int*   x       = (const int*)d_in[0];
  const int*   targets = (const int*)d_in[1];
  const float* wte     = (const float*)d_in[2];
  const float* wpe     = (const float*)d_in[3];
  const float* ln1_w   = (const float*)d_in[4];
  const float* ln1_b   = (const float*)d_in[5];
  const float* attn_w  = (const float*)d_in[6];
  const float* attn_b  = (const float*)d_in[7];
  const float* proj_w  = (const float*)d_in[8];
  const float* proj_b  = (const float*)d_in[9];
  const float* ln2_w   = (const float*)d_in[10];
  const float* ln2_b   = (const float*)d_in[11];
  const float* fc_w    = (const float*)d_in[12];
  const float* fc_b    = (const float*)d_in[13];
  const float* out_w   = (const float*)d_in[14];
  const float* out_b   = (const float*)d_in[15];
  const float* lnf_w   = (const float*)d_in[16];
  const float* lnf_b   = (const float*)d_in[17];

  float* logits = (float*)d_out;             // [T][V]
  float* loss   = logits + (size_t)Tc * Vc;  // scalar

  // workspace layout
  float*  ws   = (float*)d_ws;
  float*  h    = ws;                                  // T*D f32
  float*  nll  = h + (size_t)Tc * Dc;                 // T
  float*  vld  = nll + Tc;                            // T
  __bf16* lnb  = (__bf16*)(vld + Tc);                 // T*D
  __bf16* qkvb = lnb + (size_t)Tc * Dc;               // T*3D
  __bf16* attb = qkvb + (size_t)Tc * D3c;             // T*D
  __bf16* fcbb = attb + (size_t)Tc * Dc;              // T*4D
  __bf16* wteb = fcbb + (size_t)Tc * D4c;             // V*D
  __bf16* wT   = wteb + (size_t)Vc * Dc;              // transposed bf16 weights
  const size_t perL = (size_t)D3c * Dc + (size_t)Dc * Dc +
                      (size_t)D4c * Dc + (size_t)Dc * D4c;

  // --- one-pass weight preprocessing (bf16, [N][K]) ---
  {
    int n = Vc * Dc;
    cvt_bf16_kernel<<<(n + 255) / 256, 256, 0, stream>>>(wte, wteb, n);
  }
  for (int l = 0; l < Lc; ++l) {
    __bf16* aT = wT + l * perL;
    __bf16* pT = aT + (size_t)D3c * Dc;
    __bf16* fT = pT + (size_t)Dc * Dc;
    __bf16* oT = fT + (size_t)D4c * Dc;
    transpose_bf16_kernel<<<dim3(D3c / 32, Dc / 32), 256, 0, stream>>>(
        attn_w + (size_t)l * Dc * D3c, aT, Dc, D3c);
    transpose_bf16_kernel<<<dim3(Dc / 32, Dc / 32), 256, 0, stream>>>(
        proj_w + (size_t)l * Dc * Dc, pT, Dc, Dc);
    transpose_bf16_kernel<<<dim3(D4c / 32, Dc / 32), 256, 0, stream>>>(
        fc_w + (size_t)l * Dc * D4c, fT, Dc, D4c);
    transpose_bf16_kernel<<<dim3(Dc / 32, D4c / 32), 256, 0, stream>>>(
        out_w + (size_t)l * D4c * Dc, oT, D4c, Dc);
  }

  embed_kernel<<<Tc, 256, 0, stream>>>(x, wte, wpe, h);

  for (int l = 0; l < Lc; ++l) {
    __bf16* aT = wT + l * perL;
    __bf16* pT = aT + (size_t)D3c * Dc;
    __bf16* fT = pT + (size_t)Dc * Dc;
    __bf16* oT = fT + (size_t)D4c * Dc;

    layernorm_kernel<<<Tc, 256, 0, stream>>>(h, ln1_w + l * Dc, ln1_b + l * Dc, lnb);
    gemm_wmma<false, false, true><<<gemm_grid(Tc, D3c), 256, 0, stream>>>(
        lnb, aT, attn_b + (size_t)l * D3c, nullptr, qkvb, Tc, D3c, Dc, Dc, D3c);
    attention_kernel<<<dim3(Tc / 16, Hc), 32, 0, stream>>>(qkvb, attb);
    gemm_wmma<false, true, false><<<gemm_grid(Tc, Dc), 256, 0, stream>>>(
        attb, pT, proj_b + (size_t)l * Dc, h, h, Tc, Dc, Dc, Dc, Dc);
    layernorm_kernel<<<Tc, 256, 0, stream>>>(h, ln2_w + l * Dc, ln2_b + l * Dc, lnb);
    gemm_wmma<true, false, true><<<gemm_grid(Tc, D4c), 256, 0, stream>>>(
        lnb, fT, fc_b + (size_t)l * D4c, nullptr, fcbb, Tc, D4c, Dc, Dc, D4c);
    gemm_wmma<false, true, false><<<gemm_grid(Tc, Dc), 256, 0, stream>>>(
        fcbb, oT, out_b + (size_t)l * Dc, h, h, Tc, Dc, D4c, D4c, Dc);
  }

  layernorm_kernel<<<Tc, 256, 0, stream>>>(h, lnf_w, lnf_b, lnb);
  gemm_wmma<false, false, false><<<gemm_grid(Tc, Vc), 256, 0, stream>>>(
      lnb, wteb, nullptr, nullptr, logits, Tc, Vc, Dc, Dc, Vc);

  nll_kernel<<<Tc, 256, 0, stream>>>(logits, targets, nll, vld);
  loss_reduce_kernel<<<1, 256, 0, stream>>>(nll, vld, loss);
}
